// PCNN_ATT_24378234372239
// MI455X (gfx1250) — matrix operand order repaired and checked by reading, not compiled, not run on hardware
//
#include <hip/hip_runtime.h>
#include <hip/hip_bf16.h>
#include <math.h>

// ---------------- problem constants ----------------
#define Bb   256
#define Ss   512
#define Ee   256
#define PDd  32
#define FEATc 320          // E + 2*PD
#define FNc  230
#define TAGc 53
#define NPAD 240           // FN padded to 15 tiles of 16
#define NTILES 15
#define KTOT 960           // 3 * FEAT
#define KSTEPS 30          // 960 / 32
#define LDSROW 328         // 320 + 8 pad (f16 elems); 656B row, 16B aligned

typedef __attribute__((ext_vector_type(8)))  _Float16 v8h;
typedef __attribute__((ext_vector_type(16))) _Float16 v16h;
typedef __attribute__((ext_vector_type(8)))  float    v8f;

union AFrag { v16h v; v8h h[2]; };

// workspace layout (bytes)
#define OFF_W      0                         // packed f16 weights: 15*30*32*16 = 230400 f16
#define OFF_PCNN   460800                    // 256*3*240 f32 = 737280 B
#define OFF_BOUNDS (460800 + 737280)         // 256 * int4

// ---------------- weight pack: conv_w (3,320,230) f32 -> WMMA B frags f16 ----
__global__ void packWK(const float* __restrict__ conv_w, _Float16* __restrict__ pw) {
    int blk = blockIdx.x;            // 0..449 : nt*30 + ks
    int nt  = blk / KSTEPS;
    int ks  = blk % KSTEPS;
    int tid = threadIdx.x;           // 256 threads: lane = tid>>3, dword = tid&7
    int lane = tid >> 3;
    int dw   = tid & 7;
    int n = nt * 16 + (lane & 15);
#pragma unroll
    for (int e = 0; e < 2; ++e) {
        int j = 2 * dw + e;                       // 0..15 within lane
        int klocal = ((lane >> 4) << 3) + (j < 8 ? j : j + 8);  // lanes 0-15: K 0..7,16..23; lanes 16-31: K 8..15,24..31
        int k = ks * 32 + klocal;                 // global K in [0,960)
        int dt = k / FEATc;
        int c  = k % FEATc;
        float v = (n < FNc) ? conv_w[(dt * FEATc + c) * FNc + n] : 0.0f;
        pw[((size_t)blk * 32 + lane) * 16 + j] = (_Float16)v;
    }
}

// ---------------- zero the pooled-max buffer ----------------
__global__ void zeroK(float* __restrict__ p, int n) {
    int i = blockIdx.x * blockDim.x + threadIdx.x;
    if (i < n) p[i] = 0.0f;
}

// ---------------- per-batch segment boundaries ----------------
__global__ void boundsK(const int* __restrict__ pos1, const int* __restrict__ pos2,
                        int4* __restrict__ bounds) {
    int b = threadIdx.x;
    if (b >= Bb) return;
    int p1min = Ss, p1max = -1, p2min = Ss, p2max = -1;
    for (int i = 0; i < Ss; ++i) {
        if (pos1[b * Ss + i] == 60) { if (i < p1min) p1min = i; if (i > p1max) p1max = i; }
        if (pos2[b * Ss + i] == 60) { if (i < p2min) p2min = i; if (i > p2max) p2max = i; }
    }
    bool first = p1min < p2min;
    int4 r;
    r.x = first ? p1min : p2min;   // m1
    r.y = first ? p1max : p2max;   // m2
    r.z = first ? p2min : p1min;   // m3
    r.w = first ? p2max : p1max;   // m4
    bounds[b] = r;
}

// ---------------- fused gather + conv(WMMA) + ReLU + piecewise max ----------------
__global__ __launch_bounds__(256)
void convK(const int* __restrict__ sentence, const int* __restrict__ pos1,
           const int* __restrict__ pos2,
           const float* __restrict__ word_emb, const float* __restrict__ pos1_emb,
           const float* __restrict__ pos2_emb, const float* __restrict__ conv_b,
           const _Float16* __restrict__ pw, const int4* __restrict__ bounds,
           unsigned int* __restrict__ pcnnU) {
    __shared__ _Float16 sEmb[18 * LDSROW];     // rows t0-1 .. t0+16, f16, padded

    const int t0 = blockIdx.x * 16;
    const int b  = blockIdx.y;
    const int tid  = threadIdx.x;
    const int lane = tid & 31;
    const int wave = tid >> 5;

    // ---- gather embeds into LDS (f16) ----
    for (int idx = tid; idx < 18 * FEATc; idx += 256) {
        int row = idx / FEATc;
        int c   = idx - row * FEATc;
        int t   = t0 - 1 + row;
        float v = 0.0f;
        if (t >= 0 && t < Ss) {
            if (c < Ee) {
                int w = sentence[b * Ss + t];
                v = word_emb[(size_t)w * Ee + c];
            } else if (c < Ee + PDd) {
                int p = pos1[b * Ss + t];
                v = pos1_emb[p * PDd + (c - Ee)];
            } else {
                int p = pos2[b * Ss + t];
                v = pos2_emb[p * PDd + (c - Ee - PDd)];
            }
        }
        sEmb[row * LDSROW + c] = (_Float16)v;
    }
    __syncthreads();

    const int4 mb = bounds[b];
    const int m2 = mb.y, m4 = mb.w;
    const int segA_hi = (mb.x > 1) ? mb.x : 1;              // t <  segA_hi        -> seg 0
    const int segB_hi = (mb.z > m2 + 1) ? mb.z : m2 + 1;    // m2 <= t < segB_hi   -> seg 1

    const int mrow = lane & 15;
    const int koff = (lane >> 4) << 3;                // 0 or 8 (element offset in K-block)

    // each wave owns two consecutive n-tiles: {2w, 2w+1}; wave 7's second is a dup
    const int nt0 = 2 * wave;
    const int nt1 = (2 * wave + 1 < NTILES) ? 2 * wave + 1 : NTILES - 1;
    const bool has1 = (2 * wave + 1 < NTILES);

    const v16h* wp0 = ((const v16h*)pw) + ((size_t)nt0 * KSTEPS) * 32 + lane;
    const v16h* wp1 = ((const v16h*)pw) + ((size_t)nt1 * KSTEPS) * 32 + lane;

    v8f acc0 = {}, acc1 = {};
    v16h b0 = wp0[0];
    v16h b1 = wp1[0];

#pragma unroll
    for (int dtp1 = 0; dtp1 < 3; ++dtp1) {
#pragma unroll
        for (int kc = 0; kc < 10; ++kc) {
            const int ks = dtp1 * 10 + kc;
            // prefetch next step's B fragments (double buffer)
            v16h b0n = b0, b1n = b1;
            if (ks + 1 < KSTEPS) {
                b0n = wp0[(ks + 1) * 32];
                b1n = wp1[(ks + 1) * 32];
            }
            // A fragment from LDS (shared by both n-tiles)
            const _Float16* ab = &sEmb[(mrow + dtp1) * LDSROW + kc * 32 + koff];
            AFrag af;
            af.h[0] = *(const v8h*)(ab);        // K[koff .. koff+8)
            af.h[1] = *(const v8h*)(ab + 16);   // K[koff+16 .. koff+24)
            acc0 = __builtin_amdgcn_wmma_f32_16x16x32_f16(
                      false, af.v, false, b0, (short)0, acc0, false, false);
            acc1 = __builtin_amdgcn_wmma_f32_16x16x32_f16(
                      false, af.v, false, b1, (short)0, acc1, false, false);
            b0 = b0n; b1 = b1n;
        }
    }

    // ---- epilogue: bias + ReLU + per-segment max ----
#pragma unroll
    for (int which = 0; which < 2; ++which) {
        if (which == 1 && !has1) break;
        const v8f& acc = (which == 0) ? acc0 : acc1;
        const int nt = (which == 0) ? nt0 : nt1;
        const int n = nt * 16 + mrow;
        const float bias = (n < FNc) ? conv_b[n] : 0.0f;
        float smax0 = 0.0f, smax1 = 0.0f, smax2 = 0.0f;
#pragma unroll
        for (int r = 0; r < 8; ++r) {
            int t = t0 + r + ((lane >> 4) << 3);
            float v = acc[r] + bias;
            v = v > 0.0f ? v : 0.0f;
            if (t < segA_hi)             smax0 = fmaxf(smax0, v);
            if (t >= m2 && t < segB_hi)  smax1 = fmaxf(smax1, v);
            if (t >= m4)                 smax2 = fmaxf(smax2, v);
        }
        // fuse lane L with L^16 (same n, other 8 rows)
        smax0 = fmaxf(smax0, __shfl_xor(smax0, 16, 32));
        smax1 = fmaxf(smax1, __shfl_xor(smax1, 16, 32));
        smax2 = fmaxf(smax2, __shfl_xor(smax2, 16, 32));
        if (lane < 16 && n < FNc) {
            unsigned int* base = pcnnU + (size_t)b * 3 * NPAD + n;
            atomicMax(base,             __float_as_uint(smax0));
            atomicMax(base + NPAD,      __float_as_uint(smax1));
            atomicMax(base + 2 * NPAD,  __float_as_uint(smax2));
        }
    }
}

// ---------------- per-batch attention + relation softmax ----------------
__global__ __launch_bounds__(256)
void finalK(const float* __restrict__ pcnn, const float* __restrict__ att_weight,
            const float* __restrict__ rel_emb, const float* __restrict__ relation_bias,
            float* __restrict__ out) {
    __shared__ float s_red[256];
    __shared__ float s_att[FNc];
    __shared__ float s_a[4];

    const int b = blockIdx.x;
    const int f = threadIdx.x;
    const bool act = (f < FNc);

    float p0 = 0.f, p1 = 0.f, p2 = 0.f, aw = 0.f;
    if (act) {
        const float* pb = pcnn + (size_t)b * 3 * NPAD;
        p0 = pb[f]; p1 = pb[NPAD + f]; p2 = pb[2 * NPAD + f];
        aw = att_weight[(size_t)b * FNc + f];
    }

    // scores[s] = sum_f aw * tanh(p[s])
    float contrib[3] = { act ? aw * tanhf(p0) : 0.f,
                         act ? aw * tanhf(p1) : 0.f,
                         act ? aw * tanhf(p2) : 0.f };
#pragma unroll
    for (int s = 0; s < 3; ++s) {
        s_red[f] = contrib[s];
        __syncthreads();
        for (int off = 128; off > 0; off >>= 1) {
            if (f < off) s_red[f] += s_red[f + off];
            __syncthreads();
        }
        if (f == 0) s_a[s] = s_red[0];
        __syncthreads();
    }
    if (f == 0) {  // softmax over 3
        float mx = fmaxf(s_a[0], fmaxf(s_a[1], s_a[2]));
        float e0 = expf(s_a[0] - mx), e1 = expf(s_a[1] - mx), e2 = expf(s_a[2] - mx);
        float inv = 1.0f / (e0 + e1 + e2);
        s_a[0] = e0 * inv; s_a[1] = e1 * inv; s_a[2] = e2 * inv;
    }
    __syncthreads();
    if (act) s_att[f] = tanhf(p0 * s_a[0] + p1 * s_a[1] + p2 * s_a[2]);
    __syncthreads();

    // wei[tag] = rel_emb[tag,:] . att_out + bias
    if (f < TAGc) {
        const float* re = rel_emb + (size_t)f * FNc;
        float acc = 0.0f;
        for (int k = 0; k < FNc; ++k) acc += re[k] * s_att[k];
        s_red[f] = acc + relation_bias[(size_t)b * TAGc + f];
    }
    __syncthreads();
    if (f == 0) {  // softmax over 53
        float mx = -1e30f;
        for (int t = 0; t < TAGc; ++t) mx = fmaxf(mx, s_red[t]);
        float sum = 0.0f;
        for (int t = 0; t < TAGc; ++t) sum += expf(s_red[t] - mx);
        s_a[0] = mx; s_a[1] = 1.0f / sum;
    }
    __syncthreads();
    if (f < TAGc)
        out[(size_t)b * TAGc + f] = expf(s_red[f] - s_a[0]) * s_a[1];
}

// ---------------- launcher ----------------
extern "C" void kernel_launch(void* const* d_in, const int* in_sizes, int n_in,
                              void* d_out, int out_size, void* d_ws, size_t ws_size,
                              hipStream_t stream) {
    const int*   sentence      = (const int*)  d_in[0];
    const int*   pos1          = (const int*)  d_in[1];
    const int*   pos2          = (const int*)  d_in[2];
    const float* word_emb      = (const float*)d_in[3];
    const float* pos1_emb      = (const float*)d_in[4];
    const float* pos2_emb      = (const float*)d_in[5];
    const float* rel_emb       = (const float*)d_in[6];
    const float* conv_w        = (const float*)d_in[7];
    const float* conv_b        = (const float*)d_in[8];
    const float* att_weight    = (const float*)d_in[9];
    const float* relation_bias = (const float*)d_in[10];
    float* out = (float*)d_out;

    char* ws = (char*)d_ws;
    _Float16*     pw     = (_Float16*)    (ws + OFF_W);
    float*        pcnn   = (float*)       (ws + OFF_PCNN);
    unsigned int* pcnnU  = (unsigned int*)(ws + OFF_PCNN);
    int4*         bounds = (int4*)        (ws + OFF_BOUNDS);

    // 1) pack weights to f16 WMMA-B fragments
    packWK<<<NTILES * KSTEPS, 256, 0, stream>>>(conv_w, pw);
    // 2) zero pooled-max buffer (ReLU output >= 0, uint-bit atomicMax is exact)
    const int pcnnN = Bb * 3 * NPAD;
    zeroK<<<(pcnnN + 255) / 256, 256, 0, stream>>>(pcnn, pcnnN);
    // 3) entity segment boundaries
    boundsK<<<1, 256, 0, stream>>>(pos1, pos2, bounds);
    // 4) fused gather + WMMA conv + ReLU + piecewise max
    dim3 grid(Ss / 16, Bb);
    convK<<<grid, 256, 0, stream>>>(sentence, pos1, pos2, word_emb, pos1_emb,
                                    pos2_emb, conv_b, pw, bounds, pcnnU);
    // 5) attention + relation softmax
    finalK<<<Bb, 256, 0, stream>>>(pcnn, att_weight, rel_emb, relation_bias, out);
}
